// MatchModel_44298292691000
// MI455X (gfx1250) — compile-verified, matrix operand order checked by reading, not run on hardware
//
#include <hip/hip_runtime.h>
#include <math.h>

typedef __attribute__((ext_vector_type(16))) _Float16 v16h;
typedef __attribute__((ext_vector_type(8)))  _Float16 v8h;
typedef __attribute__((ext_vector_type(4)))  _Float16 v4h;
typedef __attribute__((ext_vector_type(8)))  float    v8f;

#define Dn   128
#define Bn   256
#define Tn   20
#define Nn   100000
#define En   800000
#define KMAX 768
#define KPAD 776   // row stride in halfs (1552 B, 16B-aligned)

// ---------------------------------------------------------------- utilities

__global__ void fill4_kernel(float* __restrict__ p, float v, size_t n4) {
    size_t i = (size_t)blockIdx.x * blockDim.x + threadIdx.x;
    if (i < n4) {
        float4 f = make_float4(v, v, v, v);
        ((float4*)p)[i] = f;
    }
}

__global__ void copy_kernel(const float* __restrict__ s, float* __restrict__ d, size_t n) {
    size_t i = (size_t)blockIdx.x * blockDim.x + threadIdx.x;
    if (i < n) d[i] = s[i];
}

__global__ void vecadd_kernel(const float* __restrict__ a, const float* __restrict__ b,
                              float* __restrict__ o, int n) {
    int i = blockIdx.x * blockDim.x + threadIdx.x;
    if (i < n) o[i] = a[i] + b[i];
}

// pack weight into f16 [N,K] layout: out[n*K+k] = trans ? W[k*N+n] : W[n*K+k]
__global__ void pack_b_kernel(const float* __restrict__ W, _Float16* __restrict__ out,
                              int Nc, int K, int trans) {
    int idx = blockIdx.x * blockDim.x + threadIdx.x;
    if (idx >= Nc * K) return;
    int n = idx / K;
    int k = idx - n * K;
    float v = trans ? W[(size_t)k * Nc + n] : W[idx];
    out[idx] = (_Float16)v;
}

// out[row] = emb[ids[row]]  (rows of D=128 floats, float4 vectorized)
__global__ void gather_rows_kernel(const float* __restrict__ emb, const int* __restrict__ ids,
                                   float* __restrict__ out, int nrows) {
    size_t idx = (size_t)blockIdx.x * blockDim.x + threadIdx.x;   // nrows*32 threads
    int row = (int)(idx >> 5);
    int c   = ((int)idx & 31) * 4;
    if (row >= nrows) return;
    const float4 v = *(const float4*)(emb + (size_t)ids[row] * Dn + c);
    *(float4*)(out + (size_t)row * Dn + c) = v;
}

// ---------------------------------------------------------------- graph ops

__global__ void degree_kernel(const int* __restrict__ src, const int* __restrict__ dst,
                              float* __restrict__ deg_o, float* __restrict__ deg_i, int E) {
    int e = blockIdx.x * blockDim.x + threadIdx.x;
    if (e < E) {
        atomicAdd(&deg_o[src[e]], 1.0f);
        atomicAdd(&deg_i[dst[e]], 1.0f);
    }
}

__global__ void rsqrt_kernel(float* __restrict__ d, int n) {
    int i = blockIdx.x * blockDim.x + threadIdx.x;
    if (i < n) d[i] = rsqrtf(fmaxf(d[i], 1.0f));
}

// agg[dst] += h[src] * rs_o[src]   (one thread per (edge, 4-float chunk))
__global__ void scatter_kernel(const float* __restrict__ h, const float* __restrict__ rs_o,
                               const int* __restrict__ src, const int* __restrict__ dst,
                               float* __restrict__ agg, int E) {
    size_t idx = (size_t)blockIdx.x * blockDim.x + threadIdx.x;   // E*32 threads
    int e = (int)(idx >> 5);
    int c = ((int)idx & 31) * 4;
    if (e >= E) return;
    int s = src[e], d = dst[e];
    float sc = rs_o[s];
    const float4 hv = *(const float4*)(h + (size_t)s * Dn + c);
    float* ap = agg + (size_t)d * Dn + c;
    atomicAdd(ap + 0, hv.x * sc);
    atomicAdd(ap + 1, hv.y * sc);
    atomicAdd(ap + 2, hv.z * sc);
    atomicAdd(ap + 3, hv.w * sc);
}

// mean readout: hg[gid[row]] += h[row]; cnt[gid[row]] += 1
__global__ void readout_kernel(const float* __restrict__ h, const int* __restrict__ gid,
                               float* __restrict__ hg, float* __restrict__ cnt, int nrows) {
    size_t idx = (size_t)blockIdx.x * blockDim.x + threadIdx.x;   // nrows*32 threads
    int row = (int)(idx >> 5);
    int c   = ((int)idx & 31) * 4;
    if (row >= nrows) return;
    int b = gid[row];
    const float4 hv = *(const float4*)(h + (size_t)row * Dn + c);
    float* hp = hg + (size_t)b * Dn + c;
    atomicAdd(hp + 0, hv.x);
    atomicAdd(hp + 1, hv.y);
    atomicAdd(hp + 2, hv.z);
    atomicAdd(hp + 3, hv.w);
    if (c == 0) atomicAdd(&cnt[b], 1.0f);
}

__global__ void meandiv_kernel(float* __restrict__ hg, const float* __restrict__ cnt) {
    int i = blockIdx.x * blockDim.x + threadIdx.x;   // B*D threads
    int b = i >> 7;
    hg[i] = hg[i] / fmaxf(cnt[b], 1.0f);
}

// ---------------------------------------------------------------- WMMA GEMM
// C[M x Ncols] = act( (HAS_RS ? rs[m] : 1) * A[M x K] @ Bh^T + bias )
// A: f32 row-major, stride lda.  Bh: f16 packed [Ncols, K] (row n = column n of B).
// grid.x = M/16, grid.y = Ncols/128, 256 threads = 8 waves; wave w owns cols [w*16,w*16+16).
// The whole 16 x K A-strip is staged to LDS once; the K loop is barrier-free.
template <bool HAS_RS>
__global__ __launch_bounds__(256) void gemm_wmma_kernel(
    const float* __restrict__ A, int lda, const float* __restrict__ rs,
    const _Float16* __restrict__ Bh,
    const float* __restrict__ bias, float* __restrict__ C, int ldc,
    int K, int act)
{
    __shared__ _Float16 a_sm[16][KPAD];      // 16 x K (K <= 768), rows 16B-aligned
    const int tid  = threadIdx.x;
    const int wave = tid >> 5;
    const int lane = tid & 31;
    const int sub  = lane >> 4;              // 0 or 1
    const int l16  = lane & 15;
    const int row0 = blockIdx.x * 16;
    const int col0 = blockIdx.y * 128 + wave * 16;
    const int n    = col0 + l16;
    const int kb   = sub * 8;

    // stage full A strip (f32 -> f16): thread = (row = tid>>4, k offset = (tid&15)*4, stride 64)
    {
        const int r   = tid >> 4;
        const int kks = (tid & 15) * 4;
        const float sc = HAS_RS ? rs[row0 + r] : 1.0f;
        const float* arow = A + (size_t)(row0 + r) * lda;
        for (int kk = kks; kk < K; kk += 64) {
            const float4 v = *(const float4*)(arow + kk);
            v4h hv;
            hv[0] = (_Float16)(v.x * sc);
            hv[1] = (_Float16)(v.y * sc);
            hv[2] = (_Float16)(v.z * sc);
            hv[3] = (_Float16)(v.w * sc);
            *(v4h*)&a_sm[r][kk] = hv;
        }
    }
    __syncthreads();

    v8f acc = {0.f, 0.f, 0.f, 0.f, 0.f, 0.f, 0.f, 0.f};
    const _Float16* brow = Bh + (size_t)n * K;

    for (int k0 = 0; k0 < K; k0 += 32) {
        const v8h alo = *(const v8h*)&a_sm[l16][k0 + kb];
        const v8h ahi = *(const v8h*)&a_sm[l16][k0 + kb + 16];
        const v8h blo = *(const v8h*)(brow + k0 + kb);
        const v8h bhi = *(const v8h*)(brow + k0 + kb + 16);

        v16h af, bfv;
        #pragma unroll
        for (int i = 0; i < 8; ++i) {
            af[i]      = alo[i];
            af[8 + i]  = ahi[i];
            bfv[i]     = blo[i];
            bfv[8 + i] = bhi[i];
        }
        acc = __builtin_amdgcn_wmma_f32_16x16x32_f16(false, af, false, bfv,
                                                     (short)0, acc, false, false);
    }

    float bv = bias ? bias[n] : 0.0f;
    #pragma unroll
    for (int r = 0; r < 8; ++r) {
        float v = acc[r] + bv;
        if (act == 1)      v = (v > 0.0f) ? v : 0.01f * v;   // leaky relu
        else if (act == 2) v = fmaxf(v, 0.0f);               // relu
        C[(size_t)(row0 + r + sub * 8) * ldc + n] = v;
    }
}

// ---------------------------------------------------------------- LSTM / path

__device__ __forceinline__ float sigf(float x) { return 1.0f / (1.0f + expf(-x)); }

__global__ void lstm_point_kernel(const float* __restrict__ Xg, int t,
                                  const float* __restrict__ G,
                                  float* __restrict__ h, float* __restrict__ c,
                                  float* __restrict__ Hs) {
    int idx = blockIdx.x * blockDim.x + threadIdx.x;   // B*D threads
    int b = idx >> 7;
    int d = idx & 127;
    const float* xg = Xg + ((size_t)b * Tn + t) * (4 * Dn);
    const float* g  = G  + (size_t)b * (4 * Dn);
    float ig = sigf (xg[d]            + g[d]);
    float fg = sigf (xg[Dn + d]       + g[Dn + d]);
    float gg = tanhf(xg[2 * Dn + d]   + g[2 * Dn + d]);
    float og = sigf (xg[3 * Dn + d]   + g[3 * Dn + d]);
    float cn = fg * c[idx] + ig * gg;
    float hn = og * tanhf(cn);
    c[idx] = cn;
    h[idx] = hn;
    Hs[((size_t)b * Tn + t) * Dn + d] = hn;
}

__global__ void pathmax_kernel(const float* __restrict__ cvec, const float* __restrict__ Hs,
                               const int* __restrict__ lens, int col, float* __restrict__ hp) {
    int idx = blockIdx.x * blockDim.x + threadIdx.x;   // B*D threads
    int b = idx >> 7;
    int d = idx & 127;
    int L = lens[b * 2 + col];
    float cv = cvec[idx];
    float m = -3.4e38f;
    #pragma unroll
    for (int t = 0; t < Tn; ++t) {
        float v = (t < L) ? cv * Hs[((size_t)b * Tn + t) * Dn + d] : 0.0f;
        m = fmaxf(m, v);
    }
    hp[idx] = m;
}

// ---------------------------------------------------------------- final head

__global__ void build_e_kernel(const int* __restrict__ u, const int* __restrict__ v,
                               const float* __restrict__ emb,
                               const float* __restrict__ hgu, const float* __restrict__ hpu,
                               const float* __restrict__ hgv, const float* __restrict__ hpv,
                               float* __restrict__ e) {
    int idx = blockIdx.x * blockDim.x + threadIdx.x;   // B*768 threads
    int b = idx / 768;
    int j = idx % 768;
    int seg = j >> 7;
    int d = j & 127;
    float val;
    switch (seg) {
        case 0: val = emb[(size_t)u[b] * Dn + d]; break;
        case 1: val = hgu[b * Dn + d];            break;
        case 2: val = hpu[b * Dn + d];            break;
        case 3: val = emb[(size_t)v[b] * Dn + d]; break;
        case 4: val = hgv[b * Dn + d];            break;
        default: val = hpv[b * Dn + d];           break;
    }
    e[(size_t)b * 768 + j] = val;
}

__global__ void score_kernel(const float* __restrict__ tmp, const float* __restrict__ emb,
                             const int* __restrict__ q, const float* __restrict__ bim_b,
                             float* __restrict__ out) {
    __shared__ float red[128];
    int b = blockIdx.x, d = threadIdx.x;
    red[d] = tmp[b * Dn + d] * emb[(size_t)q[b] * Dn + d];
    __syncthreads();
    for (int s = 64; s > 0; s >>= 1) {
        if (d < s) red[d] += red[d + s];
        __syncthreads();
    }
    if (d == 0) out[b] = red[0] + bim_b[0];
}

// ---------------------------------------------------------------- host driver

extern "C" void kernel_launch(void* const* d_in, const int* in_sizes, int n_in,
                              void* d_out, int out_size, void* d_ws, size_t ws_size,
                              hipStream_t stream) {
    (void)in_sizes; (void)n_in; (void)out_size; (void)ws_size;

    const int*   q      = (const int*)  d_in[0];
    const int*   u      = (const int*)  d_in[1];
    const int*   v      = (const int*)  d_in[2];
    const int*   gu_ids = (const int*)  d_in[3];
    const int*   gu_src = (const int*)  d_in[4];
    const int*   gu_dst = (const int*)  d_in[5];
    const int*   gu_gid = (const int*)  d_in[6];
    const int*   gv_ids = (const int*)  d_in[7];
    const int*   gv_src = (const int*)  d_in[8];
    const int*   gv_dst = (const int*)  d_in[9];
    const int*   gv_gid = (const int*)  d_in[10];
    const int*   pu     = (const int*)  d_in[11];
    const int*   pv     = (const int*)  d_in[12];
    const int*   lens   = (const int*)  d_in[13];
    const float* emb    = (const float*)d_in[14];
    const float* h0p    = (const float*)d_in[15];
    const float* c0p    = (const float*)d_in[16];
    const float* h0c    = (const float*)d_in[17];
    const float* c0c    = (const float*)d_in[18];
    const float* gcnp_w1 = (const float*)d_in[19];
    const float* gcnp_b1 = (const float*)d_in[20];
    const float* gcnp_w2 = (const float*)d_in[21];
    const float* gcnp_b2 = (const float*)d_in[22];
    const float* gcnc_w1 = (const float*)d_in[23];
    const float* gcnc_b1 = (const float*)d_in[24];
    const float* gcnc_w2 = (const float*)d_in[25];
    const float* gcnc_b2 = (const float*)d_in[26];
    const float* lstmp_wih = (const float*)d_in[27];
    const float* lstmp_whh = (const float*)d_in[28];
    const float* lstmp_bih = (const float*)d_in[29];
    const float* lstmp_bhh = (const float*)d_in[30];
    const float* lstmc_wih = (const float*)d_in[31];
    const float* lstmc_whh = (const float*)d_in[32];
    const float* lstmc_bih = (const float*)d_in[33];
    const float* lstmc_bhh = (const float*)d_in[34];
    const float* ctrlp_w = (const float*)d_in[35];
    const float* ctrlc_w = (const float*)d_in[36];
    const float* bim_w   = (const float*)d_in[37];
    const float* bim_b   = (const float*)d_in[38];

    float* ws = (float*)d_ws;
    size_t o = 0;
    float* nodeA = ws + o; o += (size_t)Nn * Dn;      // gathered / hidden features
    float* nodeB = ws + o; o += (size_t)Nn * Dn;      // aggregation buffer
    float* deg   = ws + o; o += 2 * (size_t)Nn;       // deg_o | deg_i -> rsqrt in place
    float* cnt   = ws + o; o += Bn;
    float* hgu   = ws + o; o += Bn * Dn;
    float* hgv   = ws + o; o += Bn * Dn;
    float* hpu   = ws + o; o += Bn * Dn;
    float* hpv   = ws + o; o += Bn * Dn;
    float* pembU = ws + o; o += (size_t)Bn * Tn * Dn;
    float* pembV = ws + o; o += (size_t)Bn * Tn * Dn;
    float* XgU   = ws + o; o += (size_t)Bn * Tn * 4 * Dn;
    float* XgV   = ws + o; o += (size_t)Bn * Tn * 4 * Dn;
    float* HsU   = ws + o; o += (size_t)Bn * Tn * Dn;
    float* HsV   = ws + o; o += (size_t)Bn * Tn * Dn;
    float* hSU   = ws + o; o += Bn * Dn;
    float* cSU   = ws + o; o += Bn * Dn;
    float* hSV   = ws + o; o += Bn * Dn;
    float* cSV   = ws + o; o += Bn * Dn;
    float* Gbuf  = ws + o; o += Bn * 4 * Dn;
    float* cvecU = ws + o; o += Bn * Dn;
    float* cvecV = ws + o; o += Bn * Dn;
    float* biasU = ws + o; o += 4 * Dn;
    float* biasV = ws + o; o += 4 * Dn;
    float* ebuf  = ws + o; o += Bn * 768;
    float* tmpbf = ws + o; o += Bn * Dn;

    // packed f16 weight region (16B aligned: float offsets are multiples of 4)
    _Float16* hb = (_Float16*)(ws + o);
    size_t ho = 0;
    _Float16* wp1h  = hb + ho; ho += 128 * 128;
    _Float16* wp2h  = hb + ho; ho += 128 * 128;
    _Float16* wc1h  = hb + ho; ho += 128 * 128;
    _Float16* wc2h  = hb + ho; ho += 128 * 128;
    _Float16* wihUh = hb + ho; ho += 512 * 128;
    _Float16* whhUh = hb + ho; ho += 512 * 128;
    _Float16* wihVh = hb + ho; ho += 512 * 128;
    _Float16* whhVh = hb + ho; ho += 512 * 128;
    _Float16* ctUh  = hb + ho; ho += 128 * 128;
    _Float16* ctVh  = hb + ho; ho += 128 * 128;
    _Float16* bimh  = hb + ho; ho += 128 * 768;

    auto cdiv = [](size_t a, size_t b) { return (unsigned)((a + b - 1) / b); };

    // -------- pack all weights to f16 [N,K] once --------
    pack_b_kernel<<<cdiv(128 * 128, 256), 256, 0, stream>>>(gcnp_w1, wp1h, 128, 128, 1);
    pack_b_kernel<<<cdiv(128 * 128, 256), 256, 0, stream>>>(gcnp_w2, wp2h, 128, 128, 1);
    pack_b_kernel<<<cdiv(128 * 128, 256), 256, 0, stream>>>(gcnc_w1, wc1h, 128, 128, 1);
    pack_b_kernel<<<cdiv(128 * 128, 256), 256, 0, stream>>>(gcnc_w2, wc2h, 128, 128, 1);
    pack_b_kernel<<<cdiv(512 * 128, 256), 256, 0, stream>>>(lstmp_wih, wihUh, 512, 128, 0);
    pack_b_kernel<<<cdiv(512 * 128, 256), 256, 0, stream>>>(lstmp_whh, whhUh, 512, 128, 0);
    pack_b_kernel<<<cdiv(512 * 128, 256), 256, 0, stream>>>(lstmc_wih, wihVh, 512, 128, 0);
    pack_b_kernel<<<cdiv(512 * 128, 256), 256, 0, stream>>>(lstmc_whh, whhVh, 512, 128, 0);
    pack_b_kernel<<<cdiv(128 * 128, 256), 256, 0, stream>>>(ctrlp_w, ctUh, 128, 128, 0);
    pack_b_kernel<<<cdiv(128 * 128, 256), 256, 0, stream>>>(ctrlc_w, ctVh, 128, 128, 0);
    pack_b_kernel<<<cdiv(128 * 768, 256), 256, 0, stream>>>(bim_w, bimh, 128, 768, 1);

    // -------- GCN + mean readout for one graph --------
    auto run_gcn = [&](const int* ids, const int* src, const int* dst, const int* gid,
                       const _Float16* w1h, const float* b1,
                       const _Float16* w2h, const float* b2, float* hg_out) {
        fill4_kernel<<<cdiv(2 * (size_t)Nn / 4, 256), 256, 0, stream>>>(deg, 0.0f, 2 * (size_t)Nn / 4);
        degree_kernel<<<cdiv(En, 256), 256, 0, stream>>>(src, dst, deg, deg + Nn, En);
        rsqrt_kernel<<<cdiv(2 * (size_t)Nn, 256), 256, 0, stream>>>(deg, 2 * Nn);

        gather_rows_kernel<<<cdiv((size_t)Nn * 32, 256), 256, 0, stream>>>(emb, ids, nodeA, Nn);

        // conv1: nodeA -> agg(nodeB) -> h1(nodeA)  [leaky relu]
        fill4_kernel<<<cdiv((size_t)Nn * Dn / 4, 256), 256, 0, stream>>>(nodeB, 0.0f, (size_t)Nn * Dn / 4);
        scatter_kernel<<<cdiv((size_t)En * 32, 256), 256, 0, stream>>>(nodeA, deg, src, dst, nodeB, En);
        gemm_wmma_kernel<true><<<dim3(Nn / 16, 1), 256, 0, stream>>>(
            nodeB, Dn, deg + Nn, w1h, b1, nodeA, Dn, Dn, 1);

        // conv2: nodeA -> agg(nodeB) -> h2(nodeA)  [no act]
        fill4_kernel<<<cdiv((size_t)Nn * Dn / 4, 256), 256, 0, stream>>>(nodeB, 0.0f, (size_t)Nn * Dn / 4);
        scatter_kernel<<<cdiv((size_t)En * 32, 256), 256, 0, stream>>>(nodeA, deg, src, dst, nodeB, En);
        gemm_wmma_kernel<true><<<dim3(Nn / 16, 1), 256, 0, stream>>>(
            nodeB, Dn, deg + Nn, w2h, b2, nodeA, Dn, Dn, 0);

        // mean readout
        fill4_kernel<<<cdiv((size_t)Bn * Dn / 4, 256), 256, 0, stream>>>(hg_out, 0.0f, (size_t)Bn * Dn / 4);
        fill4_kernel<<<cdiv((size_t)Bn / 4, 256), 256, 0, stream>>>(cnt, 0.0f, (size_t)Bn / 4);
        readout_kernel<<<cdiv((size_t)Nn * 32, 256), 256, 0, stream>>>(nodeA, gid, hg_out, cnt, Nn);
        meandiv_kernel<<<cdiv((size_t)Bn * Dn, 256), 256, 0, stream>>>(hg_out, cnt);
    };

    run_gcn(gu_ids, gu_src, gu_dst, gu_gid, wp1h, gcnp_b1, wp2h, gcnp_b2, hgu);
    run_gcn(gv_ids, gv_src, gv_dst, gv_gid, wc1h, gcnc_b1, wc2h, gcnc_b2, hgv);

    // -------- LSTM path encoder for one side --------
    auto run_path = [&](const int* pids, int lens_col, const _Float16* ctrlh,
                        const _Float16* wihh, const _Float16* whhh,
                        const float* bih, const float* bhh,
                        const float* h0, const float* c0,
                        float* pembBuf, float* XgBuf, float* HsBuf,
                        float* hBuf, float* cBuf, float* cvecBuf, float* biasBuf,
                        float* hpOut) {
        gather_rows_kernel<<<cdiv((size_t)Bn * Tn * 32, 256), 256, 0, stream>>>(
            emb, pids, pembBuf, Bn * Tn);
        vecadd_kernel<<<cdiv(4 * Dn, 256), 256, 0, stream>>>(bih, bhh, biasBuf, 4 * Dn);

        // Xg = pemb @ wih^T + (bih+bhh)   for all timesteps at once
        gemm_wmma_kernel<false><<<dim3(Bn * Tn / 16, 4), 256, 0, stream>>>(
            pembBuf, Dn, nullptr, wihh, biasBuf, XgBuf, 4 * Dn, Dn, 0);

        // ctrl vector: relu(pemb[:,0,:] @ ctrl_w^T)
        gemm_wmma_kernel<false><<<dim3(Bn / 16, 1), 256, 0, stream>>>(
            pembBuf, Tn * Dn, nullptr, ctrlh, nullptr, cvecBuf, Dn, Dn, 2);

        copy_kernel<<<cdiv((size_t)Bn * Dn, 256), 256, 0, stream>>>(h0, hBuf, (size_t)Bn * Dn);
        copy_kernel<<<cdiv((size_t)Bn * Dn, 256), 256, 0, stream>>>(c0, cBuf, (size_t)Bn * Dn);

        for (int t = 0; t < Tn; ++t) {
            gemm_wmma_kernel<false><<<dim3(Bn / 16, 4), 256, 0, stream>>>(
                hBuf, Dn, nullptr, whhh, nullptr, Gbuf, 4 * Dn, Dn, 0);
            lstm_point_kernel<<<cdiv((size_t)Bn * Dn, 256), 256, 0, stream>>>(
                XgBuf, t, Gbuf, hBuf, cBuf, HsBuf);
        }
        pathmax_kernel<<<cdiv((size_t)Bn * Dn, 256), 256, 0, stream>>>(
            cvecBuf, HsBuf, lens, lens_col, hpOut);
    };

    run_path(pu, 0, ctUh, wihUh, whhUh, lstmp_bih, lstmp_bhh,
             h0p, c0p, pembU, XgU, HsU, hSU, cSU, cvecU, biasU, hpu);
    run_path(pv, 1, ctVh, wihVh, whhVh, lstmc_bih, lstmc_bhh,
             h0c, c0c, pembV, XgV, HsV, hSV, cSV, cvecV, biasV, hpv);

    // -------- bilinear scoring head --------
    build_e_kernel<<<cdiv((size_t)Bn * 768, 256), 256, 0, stream>>>(
        u, v, emb, hgu, hpu, hgv, hpv, ebuf);
    gemm_wmma_kernel<false><<<dim3(Bn / 16, 1), 256, 0, stream>>>(
        ebuf, 6 * Dn, nullptr, bimh, nullptr, tmpbf, Dn, 6 * Dn, 0);
    score_kernel<<<Bn, 128, 0, stream>>>(tmpbf, emb, q, bim_b, (float*)d_out);
}
